// GatedGCNLayer_47201690583086
// MI455X (gfx1250) — matrix operand order repaired and checked by reading, not compiled, not run on hardware
//
#include <hip/hip_runtime.h>
#include <hip/hip_bf16.h>

#define DIM 64

typedef __attribute__((ext_vector_type(16))) __bf16 v16bf;
typedef __attribute__((ext_vector_type(8)))  __bf16 bf16x8;
typedef __attribute__((ext_vector_type(8)))  float  v8f;

union V16 { v16bf v; bf16x8 h[2]; };

// ---------------- zero agg + stats ----------------
__global__ void k_zero(float* __restrict__ p, long n) {
    long i = (long)blockIdx.x * blockDim.x + threadIdx.x;
    if (i < n) p[i] = 0.0f;
}

// ---------------- feature f32 -> bf16 ----------------
__global__ void k_cvt_feat(const float* __restrict__ f, __bf16* __restrict__ fb, int n) {
    int i = blockIdx.x * blockDim.x + threadIdx.x;
    if (i < n) fb[i] = (__bf16)f[i];
}

// ---------------- pack W{k,q,v} into WMMA B-operand lane layout ----------------
// B operand (32x16 bf16), contiguous k order used here for A and B: VGPR v, lane = half*16+n holds
// B[half*16 + 2v + p][n]. We store lane-major so the GEMM reads 32B contiguous/lane:
//   Bswz[((tile*2 + s)*32 + lane)*16 + e] = W[kg][colg]
//   kg = s*32 + (lane>>4)*16 + e,  colg = tile*16 + (lane&15)
__global__ void k_pack_B(const float* __restrict__ Wk, const float* __restrict__ Wq,
                         const float* __restrict__ Wv, __bf16* __restrict__ Bswz) {
    int i = blockIdx.x * blockDim.x + threadIdx.x;   // 12*2*32*16 = 12288
    if (i >= 12 * 2 * 32 * 16) return;
    int e    = i & 15;
    int lane = (i >> 4) & 31;
    int s    = (i >> 9) & 1;
    int tile = i >> 10;
    int kg   = s * 32 + (lane >> 4) * 16 + e;
    int colg = tile * 16 + (lane & 15);
    const float* W = (colg < 64) ? Wk : (colg < 128) ? Wq : Wv;
    Bswz[i] = (__bf16)W[kg * DIM + (colg & 63)];
}

// ---------------- fused QKV GEMM: [N,64] x [64,192] via v_wmma_f32_16x16x32_bf16 ----
// 12 waves/block; wave t computes a 16x16 tile of output columns [16t,16t+16).
__global__ void __launch_bounds__(384)
k_qkv_wmma(const __bf16* __restrict__ featb, const __bf16* __restrict__ Bswz,
           const float* __restrict__ bk, const float* __restrict__ bq,
           const float* __restrict__ bv,
           float* __restrict__ K, float* __restrict__ Q, float* __restrict__ V, int N) {
    int lane = threadIdx.x & 31;
    int tile = threadIdx.x >> 5;          // 0..11
    int half = lane >> 4;
    int n    = lane & 15;
    int rowbase = blockIdx.x * 16;

    // A operand: lane (half,n) holds row m=n; VGPRs 0-3: K = s*32 + half*16 + [0..7],
    // VGPRs 4-7: K = s*32 + half*16 + 8 + [0..7]  -> two contiguous 16B loads.
    int m = rowbase + n;
    if (m >= N) m = N - 1;                 // clamp load, keep EXEC full for WMMA

    v8f c = {};
#pragma unroll
    for (int s = 0; s < 2; ++s) {
        V16 a, b;
        const __bf16* ap = featb + (long)m * DIM + s * 32 + half * 16;
        a.h[0] = *(const bf16x8*)(ap);
        a.h[1] = *(const bf16x8*)(ap + 8);
        const __bf16* bp = Bswz + ((tile * 2 + s) * 32 + lane) * 16;
        b.h[0] = *(const bf16x8*)(bp);
        b.h[1] = *(const bf16x8*)(bp + 8);
        c = __builtin_amdgcn_wmma_f32_16x16x32_bf16(
                /*neg_a=*/false, a.v, /*neg_b=*/false, b.v,
                /*c_mod=*/(short)0, c, /*reuse_a=*/false, /*reuse_b=*/false);
    }

    // C/D layout: VGPR r, lanes 0-15: M=r, lanes 16-31: M=8+r; N = lane&15.
    int colg = tile * 16 + n;
    const float* bb = (colg < 64) ? bk : (colg < 128) ? bq : bv;
    float*       O  = (colg < 64) ? K  : (colg < 128) ? Q  : V;
    int   col  = colg & 63;
    float bias = bb[col];
#pragma unroll
    for (int r = 0; r < 8; ++r) {
        int row = rowbase + half * 8 + r;
        if (row < N) O[(long)row * DIM + col] = c[r] + bias;
    }
}

// ---------------- edge phase: gather -> sigmoid gate -> scatter-add ----------------
// One wave per edge; lane handles cols {2*lane, 2*lane+1} (float2 => 256B/row coalesced).
__global__ void k_edge(const int* __restrict__ ei, const float* __restrict__ K,
                       const float* __restrict__ Q, const float* __restrict__ V,
                       float* __restrict__ agg, int E) {
    int lane = threadIdx.x & 31;
    int e = blockIdx.x * (blockDim.x >> 5) + (threadIdx.x >> 5);
    if (e >= E) return;
    int src = ei[e];
    int dst = ei[E + e];
    float2 k2 = ((const float2*)(K + (long)dst * DIM))[lane];
    float2 q2 = ((const float2*)(Q + (long)src * DIM))[lane];
    float2 v2 = ((const float2*)(V + (long)src * DIM))[lane];
    float gx = 1.0f / (1.0f + __expf(-(k2.x + q2.x)));
    float gy = 1.0f / (1.0f + __expf(-(k2.y + q2.y)));
    float* ap = agg + (long)dst * DIM + lane * 2;
    unsafeAtomicAdd(ap,     gx * v2.x);   // global_atomic_add_f32, no return
    unsafeAtomicAdd(ap + 1, gy * v2.y);
}

// ---------------- per-channel sum / sumsq of h = agg + bias ----------------
__global__ void k_stats(const float* __restrict__ agg, const float* __restrict__ bias,
                        float* __restrict__ stats, int N) {
    __shared__ float sh[2][256];
    int tid = threadIdx.x;
    int c = tid & 63, g = tid >> 6;       // 4 row-groups share a column
    float b = bias[c];
    float s = 0.0f, s2 = 0.0f;
    for (int r = blockIdx.x * 4 + g; r < N; r += gridDim.x * 4) {
        float h = agg[(long)r * DIM + c] + b;
        s += h; s2 += h * h;
    }
    sh[0][tid] = s; sh[1][tid] = s2;
    __syncthreads();
    if (tid < 64) {
        float t  = sh[0][tid] + sh[0][64 + tid] + sh[0][128 + tid] + sh[0][192 + tid];
        float t2 = sh[1][tid] + sh[1][64 + tid] + sh[1][128 + tid] + sh[1][192 + tid];
        unsafeAtomicAdd(&stats[tid],      t);
        unsafeAtomicAdd(&stats[64 + tid], t2);
    }
}

// ---------------- batchnorm (batch stats) + ReLU ----------------
__global__ void k_bn_relu(const float* __restrict__ agg, const float* __restrict__ bias,
                          const float* __restrict__ gamma, const float* __restrict__ beta,
                          const float* __restrict__ stats, float* __restrict__ out,
                          int total, float invN) {
    int i = blockIdx.x * blockDim.x + threadIdx.x;
    if (i >= total) return;
    int c = i & 63;
    float mean = stats[c] * invN;
    float var  = stats[64 + c] * invN - mean * mean;
    float rs   = rsqrtf(var + 1e-5f);
    float h    = agg[i] + bias[c];
    float y    = (h - mean) * rs * gamma[c] + beta[c];
    out[i] = fmaxf(y, 0.0f);
}

extern "C" void kernel_launch(void* const* d_in, const int* in_sizes, int n_in,
                              void* d_out, int out_size, void* d_ws, size_t ws_size,
                              hipStream_t stream) {
    const float* feature = (const float*)d_in[0];
    const int*   ei      = (const int*)  d_in[1];
    const float* Wk      = (const float*)d_in[2];
    const float* bk      = (const float*)d_in[3];
    const float* Wq      = (const float*)d_in[4];
    const float* bq      = (const float*)d_in[5];
    const float* Wv      = (const float*)d_in[6];
    const float* bv      = (const float*)d_in[7];
    const float* bias    = (const float*)d_in[8];
    const float* gamma   = (const float*)d_in[9];
    const float* beta    = (const float*)d_in[10];
    float* out = (float*)d_out;

    int N = in_sizes[0] / DIM;
    int E = in_sizes[1] / 2;

    // workspace layout (all fits in ~58 MB; K/Q/V/agg are L2-resident on MI455X)
    float* K     = (float*)d_ws;
    float* Q     = K + (size_t)N * DIM;
    float* V     = Q + (size_t)N * DIM;
    float* agg   = V + (size_t)N * DIM;
    float* stats = agg + (size_t)N * DIM;          // 128 floats, contiguous after agg
    __bf16* featb = (__bf16*)(stats + 128);
    __bf16* Bswz  = featb + (size_t)N * DIM;       // 12288 bf16

    long nzero = (long)N * DIM + 128;              // agg + stats
    k_zero<<<(int)((nzero + 255) / 256), 256, 0, stream>>>(agg, nzero);

    int nd = N * DIM;
    k_cvt_feat<<<(nd + 255) / 256, 256, 0, stream>>>(feature, featb, nd);
    k_pack_B<<<(12288 + 255) / 256, 256, 0, stream>>>(Wk, Wq, Wv, Bswz);

    k_qkv_wmma<<<(N + 15) / 16, 384, 0, stream>>>(featb, Bswz, bk, bq, bv, K, Q, V, N);

    k_edge<<<(E + 7) / 8, 256, 0, stream>>>(ei, K, Q, V, agg, E);

    k_stats<<<256, 256, 0, stream>>>(agg, bias, stats, N);
    k_bn_relu<<<(nd + 255) / 256, 256, 0, stream>>>(agg, bias, gamma, beta, stats, out,
                                                    nd, 1.0f / (float)N);
}
